// LinearQwen3VLMoeTextSparseMoeBlock_20014547599395
// MI455X (gfx1250) — matrix operand order repaired and checked by reading, not compiled, not run on hardware
//
#include <hip/hip_runtime.h>
#include <hip/hip_bf16.h>

// MoE (Qwen3-VL-style sparse block): B=1, S=1024, H=2048, E=8, I=1408, top_k=2.
// router -> per-expert compaction -> gathered bf16 WMMA GEMMs with
// double-buffered LDS + register-staged software pipeline; moe_down uses
// GLOBAL_LOAD_ASYNC_TO_LDS (ASYNCcnt) for the pre-converted bf16 A tiles.
// Workspace requirement: ~22 MB (layout in kernel_launch).

#define H_DIM 2048
#define I_DIM 1408
#define E_NUM 8
#define T_NUM 1024

#define TILE_M 128
#define TILE_N 64
#define TILE_K 32
#define LDP 40   // LDS row stride (u16): 80B, 16B-aligned rows, staggers banks

typedef __attribute__((ext_vector_type(16))) __bf16          v16bf;
typedef __attribute__((ext_vector_type(8)))  float           v8f;
typedef __attribute__((ext_vector_type(8)))  unsigned short  v8u16;
typedef __attribute__((ext_vector_type(16))) unsigned short  v16u16;
typedef __attribute__((ext_vector_type(4)))  unsigned short  v4u16;

__device__ __forceinline__ unsigned short f2bf(float f) {
  unsigned int u = __float_as_uint(f);
  u += 0x7FFFu + ((u >> 16) & 1u);       // round-to-nearest-even
  return (unsigned short)(u >> 16);
}

// Build a 16-element bf16 fragment from two contiguous 16B LDS chunks.
__device__ __forceinline__ v16bf ld_frag(const unsigned short* p0,
                                         const unsigned short* p1) {
  v8u16 lo = *(const v8u16*)p0;
  v8u16 hi = *(const v8u16*)p1;
  v16u16 t;
#pragma unroll
  for (int i = 0; i < 8; ++i) { t[i] = lo[i]; t[i + 8] = hi[i]; }
  return __builtin_bit_cast(v16bf, t);
}

__device__ __forceinline__ v8f wmma_bf16(v16bf a, v16bf b, v8f c) {
  return __builtin_amdgcn_wmma_f32_16x16x32_bf16(
      false, a, false, b, (short)0, c, false, false);
}

// Async DMA: 16B global -> LDS, tracked by ASYNCcnt (no VGPR round-trip).
__device__ __forceinline__ void async_copy_b128(void* lds_ptr, const void* gptr) {
  unsigned lds_off = (unsigned)(unsigned long long)lds_ptr;  // addr[31:0] = LDS offset
  asm volatile("global_load_async_to_lds_b128 %0, %1, off"
               :: "v"(lds_off), "v"(gptr) : "memory");
}
__device__ __forceinline__ void wait_asynccnt0() {
  asm volatile("s_wait_asynccnt 0x0" ::: "memory");
}

// ---------------------------------------------------------------------------
__global__ void zero_counts(int* counts) {
  if (threadIdx.x < E_NUM) counts[threadIdx.x] = 0;
}

// ---------------------------------------------------------------------------
// Router: one block per token; wave e computes logit for expert e.
__global__ void router_kernel(const float* __restrict__ x,
                              const float* __restrict__ gate_w,
                              int* __restrict__ counts,
                              int* __restrict__ tok_idx,
                              int* __restrict__ slot_e,
                              int* __restrict__ slot_pos,
                              float* __restrict__ slot_w) {
  const int t   = blockIdx.x;
  const int tid = threadIdx.x;
  __shared__ float xs[H_DIM];
  __shared__ float lg[E_NUM];

#pragma unroll
  for (int r = 0; r < H_DIM / 256; ++r)
    xs[tid + r * 256] = x[(size_t)t * H_DIM + tid + r * 256];
  __syncthreads();

  const int e    = tid >> 5;
  const int lane = tid & 31;
  const float* gw = gate_w + (size_t)e * H_DIM;
  float p = 0.f;
#pragma unroll 8
  for (int j = 0; j < H_DIM / 32; ++j)
    p += xs[lane + j * 32] * gw[lane + j * 32];
#pragma unroll
  for (int off = 16; off > 0; off >>= 1)
    p += __shfl_xor(p, off, 32);
  if (lane == 0) lg[e] = p;
  __syncthreads();

  if (tid == 0) {
    float mx = lg[0];
#pragma unroll
    for (int i = 1; i < E_NUM; ++i) mx = fmaxf(mx, lg[i]);
    float pr[E_NUM]; float s = 0.f;
#pragma unroll
    for (int i = 0; i < E_NUM; ++i) { pr[i] = __expf(lg[i] - mx); s += pr[i]; }
    float inv = 1.f / s;
#pragma unroll
    for (int i = 0; i < E_NUM; ++i) pr[i] *= inv;
    int i0 = 0;
#pragma unroll
    for (int i = 1; i < E_NUM; ++i) if (pr[i] > pr[i0]) i0 = i;
    int i1 = (i0 == 0) ? 1 : 0;
#pragma unroll
    for (int i = 0; i < E_NUM; ++i)
      if (i != i0 && pr[i] > pr[i1]) i1 = i;
    float w0 = pr[i0], w1 = pr[i1];
    float wn = 1.f / (w0 + w1);
    w0 *= wn; w1 *= wn;

    int p0 = atomicAdd(&counts[i0], 1);
    int p1 = atomicAdd(&counts[i1], 1);
    tok_idx[i0 * T_NUM + p0] = t;
    tok_idx[i1 * T_NUM + p1] = t;
    slot_e[2 * t] = i0;  slot_pos[2 * t] = p0;  slot_w[2 * t] = w0;
    slot_e[2 * t + 1] = i1; slot_pos[2 * t + 1] = p1; slot_w[2 * t + 1] = w1;
  }
}

// ---------------------------------------------------------------------------
__global__ void calc_offsets(const int* __restrict__ counts,
                             int* __restrict__ offsets) {
  if (threadIdx.x == 0) {
    int a = 0;
#pragma unroll
    for (int e = 0; e < E_NUM; ++e) { offsets[e] = a; a += counts[e]; }
  }
}

// ---------------------------------------------------------------------------
// act = silu(Xg @ Wg) * (Xg @ Wu) ; gathered tokens, double-buffered bf16 WMMA.
__global__ void __launch_bounds__(256)
moe_gateup(const float* __restrict__ x,
           const float* __restrict__ w_gate,
           const float* __restrict__ w_up,
           const int* __restrict__ counts,
           const int* __restrict__ offsets,
           const int* __restrict__ tok_idx,
           unsigned short* __restrict__ act) {
  const int e     = blockIdx.z;
  const int cnt   = counts[e];
  const int mtile = blockIdx.y;
  if (mtile * TILE_M >= cnt) return;
  const int n0   = blockIdx.x * TILE_N;
  const int tid  = threadIdx.x;
  const int lane = tid & 31;
  const int wave = tid >> 5;
  const int half = lane >> 4;
  const int l15  = lane & 15;

  __shared__ __align__(16) unsigned short Xs[2][TILE_M][LDP];
  __shared__ __align__(16) unsigned short Gs[2][TILE_N][LDP];  // transposed [n][k]
  __shared__ __align__(16) unsigned short Us[2][TILE_N][LDP];
  __shared__ int toks[TILE_M];

  if (tid < TILE_M) {
    int pos = mtile * TILE_M + tid;
    toks[tid] = (pos < cnt) ? tok_idx[e * T_NUM + pos] : 0;
  }
  __syncthreads();

  const float* wg = w_gate + (size_t)e * H_DIM * I_DIM;
  const float* wu = w_up   + (size_t)e * H_DIM * I_DIM;

  // per-thread staging registers
  float4 rx[4], rg[2], ru[2];
  // X: 1024 float4 per tile -> 4/thread: m = idx>>3, k = (idx&7)*4
  // W: 512  float4 per tile -> 2/thread: k = idx>>4, n = (idx&15)*4
  const int xm[4] = { (0*256+tid) >> 3, (1*256+tid) >> 3, (2*256+tid) >> 3, (3*256+tid) >> 3 };
  const int xk    = (tid & 7) * 4;
  const int wk[2] = { (0*256+tid) >> 4, (1*256+tid) >> 4 };
  const int wn    = (tid & 15) * 4;

  auto stage_load = [&](int kk) {
#pragma unroll
    for (int i = 0; i < 4; ++i)
      rx[i] = *(const float4*)&x[(size_t)toks[xm[i]] * H_DIM + kk + xk];
#pragma unroll
    for (int i = 0; i < 2; ++i) {
      size_t g = (size_t)(kk + wk[i]) * I_DIM + n0 + wn;
      rg[i] = *(const float4*)&wg[g];
      ru[i] = *(const float4*)&wu[g];
    }
    // prefetch two tiles ahead
    if (kk + TILE_K < H_DIM && tid < TILE_K) {
      __builtin_prefetch(&wg[(size_t)(kk + TILE_K + tid) * I_DIM + n0], 0, 1);
      __builtin_prefetch(&wu[(size_t)(kk + TILE_K + tid) * I_DIM + n0], 0, 1);
    }
  };
  auto stage_store = [&](int b) {
#pragma unroll
    for (int i = 0; i < 4; ++i) {
      v4u16 pk = { f2bf(rx[i].x), f2bf(rx[i].y), f2bf(rx[i].z), f2bf(rx[i].w) };
      *(v4u16*)&Xs[b][xm[i]][xk] = pk;
    }
#pragma unroll
    for (int i = 0; i < 2; ++i) {
      Gs[b][wn + 0][wk[i]] = f2bf(rg[i].x);
      Gs[b][wn + 1][wk[i]] = f2bf(rg[i].y);
      Gs[b][wn + 2][wk[i]] = f2bf(rg[i].z);
      Gs[b][wn + 3][wk[i]] = f2bf(rg[i].w);
      Us[b][wn + 0][wk[i]] = f2bf(ru[i].x);
      Us[b][wn + 1][wk[i]] = f2bf(ru[i].y);
      Us[b][wn + 2][wk[i]] = f2bf(ru[i].z);
      Us[b][wn + 3][wk[i]] = f2bf(ru[i].w);
    }
  };

  v8f cg[4] = {};
  v8f cu[4] = {};

  // prologue: fill buffer 0
  stage_load(0);
  stage_store(0);
  __syncthreads();

  const int mrow = wave * 16 + l15;
  int buf = 0;
  for (int k0 = 0; k0 < H_DIM; k0 += TILE_K, buf ^= 1) {
    const bool has_next = (k0 + TILE_K < H_DIM);
    if (has_next) stage_load(k0 + TILE_K);   // globals in flight during WMMAs

    // preload all fragments, then back-to-back WMMAs
    v16bf a = ld_frag(&Xs[buf][mrow][half * 16], &Xs[buf][mrow][half * 16 + 8]);
    v16bf bg[4], bu[4];
#pragma unroll
    for (int s = 0; s < 4; ++s) {
      const int nrow = s * 16 + l15;
      bg[s] = ld_frag(&Gs[buf][nrow][half * 16], &Gs[buf][nrow][half * 16 + 8]);
      bu[s] = ld_frag(&Us[buf][nrow][half * 16], &Us[buf][nrow][half * 16 + 8]);
    }
#pragma unroll
    for (int s = 0; s < 4; ++s) cg[s] = wmma_bf16(a, bg[s], cg[s]);
#pragma unroll
    for (int s = 0; s < 4; ++s) cu[s] = wmma_bf16(a, bu[s], cu[s]);

    if (has_next) stage_store(buf ^ 1);
    __syncthreads();
  }

  // epilogue: act = silu(g)*u, bf16 to compacted rows
  const int roff = offsets[e];
#pragma unroll
  for (int s = 0; s < 4; ++s) {
#pragma unroll
    for (int j = 0; j < 8; ++j) {
      int m   = wave * 16 + j + 8 * half;
      int pos = mtile * TILE_M + m;
      if (pos < cnt) {
        float g = cg[s][j], u = cu[s][j];
        float a = (g / (1.f + __expf(-g))) * u;
        act[(size_t)(roff + pos) * I_DIM + n0 + s * 16 + l15] = f2bf(a);
      }
    }
  }
}

// ---------------------------------------------------------------------------
// y = act @ w_down[e] ; A tiles are bf16 already -> async DMA straight to LDS.
__global__ void __launch_bounds__(256)
moe_down(const unsigned short* __restrict__ act,
         const float* __restrict__ w_down,
         const int* __restrict__ counts,
         const int* __restrict__ offsets,
         float* __restrict__ ybuf) {
  const int e     = blockIdx.z;
  const int cnt   = counts[e];
  const int mtile = blockIdx.y;
  if (mtile * TILE_M >= cnt) return;
  const int n0   = blockIdx.x * TILE_N;
  const int tid  = threadIdx.x;
  const int lane = tid & 31;
  const int wave = tid >> 5;
  const int half = lane >> 4;
  const int l15  = lane & 15;

  __shared__ __align__(16) unsigned short As[2][TILE_M][LDP];
  __shared__ __align__(16) unsigned short Ws[2][TILE_N][LDP];

  const int rbase = offsets[e] + mtile * TILE_M;
  const float* wd = w_down + (size_t)e * I_DIM * H_DIM;

  // A: 512 16B-chunks per tile -> 2/thread: m = idx>>2, k = (idx&3)*8
  // W: 512 float4 per tile     -> 2/thread: k = idx>>4, n = (idx&15)*4
  const int am[2] = { (0*256+tid) >> 2, (1*256+tid) >> 2 };
  const int ak    = (tid & 3) * 8;
  const int wk[2] = { (0*256+tid) >> 4, (1*256+tid) >> 4 };
  const int wn    = (tid & 15) * 4;

  float4 rw[2];

  // A tile: async global->LDS DMA (ASYNCcnt), no VGPR staging, no conversion.
  auto stage_async_A = [&](int kk, int b) {
#pragma unroll
    for (int i = 0; i < 2; ++i)
      async_copy_b128(&As[b][am[i]][ak],
                      &act[(size_t)(rbase + am[i]) * I_DIM + kk + ak]);
  };
  auto stage_load_W = [&](int kk) {
#pragma unroll
    for (int i = 0; i < 2; ++i)
      rw[i] = *(const float4*)&wd[(size_t)(kk + wk[i]) * H_DIM + n0 + wn];
    if (kk + TILE_K < I_DIM && tid < TILE_K) {
      __builtin_prefetch(&wd[(size_t)(kk + TILE_K + tid) * H_DIM + n0], 0, 1);
    }
  };
  auto stage_store_W = [&](int b) {
#pragma unroll
    for (int i = 0; i < 2; ++i) {
      Ws[b][wn + 0][wk[i]] = f2bf(rw[i].x);
      Ws[b][wn + 1][wk[i]] = f2bf(rw[i].y);
      Ws[b][wn + 2][wk[i]] = f2bf(rw[i].z);
      Ws[b][wn + 3][wk[i]] = f2bf(rw[i].w);
    }
  };

  v8f cc[4] = {};

  // prologue
  stage_async_A(0, 0);
  stage_load_W(0);
  stage_store_W(0);
  wait_asynccnt0();
  __syncthreads();

  const int mrow = wave * 16 + l15;
  int buf = 0;
  for (int k0 = 0; k0 < I_DIM; k0 += TILE_K, buf ^= 1) {
    const bool has_next = (k0 + TILE_K < I_DIM);
    if (has_next) {
      stage_async_A(k0 + TILE_K, buf ^ 1);  // DMA into next buffer during WMMAs
      stage_load_W(k0 + TILE_K);
    }

    v16bf a = ld_frag(&As[buf][mrow][half * 16], &As[buf][mrow][half * 16 + 8]);
    v16bf b[4];
#pragma unroll
    for (int s = 0; s < 4; ++s) {
      const int nrow = s * 16 + l15;
      b[s] = ld_frag(&Ws[buf][nrow][half * 16], &Ws[buf][nrow][half * 16 + 8]);
    }
#pragma unroll
    for (int s = 0; s < 4; ++s) cc[s] = wmma_bf16(a, b[s], cc[s]);

    if (has_next) {
      stage_store_W(buf ^ 1);
      wait_asynccnt0();
    }
    __syncthreads();
  }

#pragma unroll
  for (int s = 0; s < 4; ++s) {
#pragma unroll
    for (int j = 0; j < 8; ++j) {
      int m   = wave * 16 + j + 8 * half;
      int pos = mtile * TILE_M + m;
      if (pos < cnt) {
        ybuf[(size_t)(rbase + m) * H_DIM + n0 + s * 16 + l15] = cc[s][j];
      }
    }
  }
}

// ---------------------------------------------------------------------------
__global__ void combine_kernel(const float* __restrict__ ybuf,
                               const int* __restrict__ offsets,
                               const int* __restrict__ slot_e,
                               const int* __restrict__ slot_pos,
                               const float* __restrict__ slot_w,
                               float* __restrict__ out) {
  const int t = blockIdx.y;
  const int h = (blockIdx.x * 256 + threadIdx.x) * 4;
  const int e0 = slot_e[2 * t], e1 = slot_e[2 * t + 1];
  const int r0 = offsets[e0] + slot_pos[2 * t];
  const int r1 = offsets[e1] + slot_pos[2 * t + 1];
  const float w0 = slot_w[2 * t], w1 = slot_w[2 * t + 1];
  float4 a = *(const float4*)&ybuf[(size_t)r0 * H_DIM + h];
  float4 b = *(const float4*)&ybuf[(size_t)r1 * H_DIM + h];
  float4 o;
  o.x = w0 * a.x + w1 * b.x;
  o.y = w0 * a.y + w1 * b.y;
  o.z = w0 * a.z + w1 * b.z;
  o.w = w0 * a.w + w1 * b.w;
  *(float4*)&out[(size_t)t * H_DIM + h] = o;
}

// ---------------------------------------------------------------------------
extern "C" void kernel_launch(void* const* d_in, const int* in_sizes, int n_in,
                              void* d_out, int out_size, void* d_ws, size_t ws_size,
                              hipStream_t stream) {
  const float* x      = (const float*)d_in[0];   // [1,1024,2048]
  const float* gate_w = (const float*)d_in[1];   // [8,2048]
  const float* w_gate = (const float*)d_in[2];   // [8,2048,1408]
  const float* w_up   = (const float*)d_in[3];   // [8,2048,1408]
  const float* w_down = (const float*)d_in[4];   // [8,1408,2048]
  float* out = (float*)d_out;                    // [1,1024,2048]

  // Workspace layout (bytes):
  //   0       counts[8]
  //   64      offsets[8]
  //   128     tok_idx[8*1024]                (32768)
  //   32896   slot_e[2048]                   (8192)
  //   41088   slot_pos[2048]                 (8192)
  //   49280   slot_w[2048]                   (8192)
  //   57600   act bf16 [(2048+128) x 1408]   (6127616)   tile-overrun pad rows
  //   6185216 ybuf f32 [2048 x 2048]         (16777216)
  //   total ~22.96 MB
  char* ws = (char*)d_ws;
  int*   counts   = (int*)(ws + 0);
  int*   offsets  = (int*)(ws + 64);
  int*   tok_idx  = (int*)(ws + 128);
  int*   slot_e   = (int*)(ws + 32896);
  int*   slot_pos = (int*)(ws + 41088);
  float* slot_w   = (float*)(ws + 49280);
  unsigned short* act = (unsigned short*)(ws + 57600);
  float* ybuf     = (float*)(ws + 6185216);

  zero_counts<<<1, 32, 0, stream>>>(counts);
  router_kernel<<<T_NUM, 256, 0, stream>>>(x, gate_w, counts, tok_idx,
                                           slot_e, slot_pos, slot_w);
  calc_offsets<<<1, 32, 0, stream>>>(counts, offsets);
  moe_gateup<<<dim3(I_DIM / TILE_N, T_NUM / TILE_M, E_NUM), 256, 0, stream>>>(
      x, w_gate, w_up, counts, offsets, tok_idx, act);
  moe_down<<<dim3(H_DIM / TILE_N, T_NUM / TILE_M, E_NUM), 256, 0, stream>>>(
      act, w_down, counts, offsets, ybuf);
  combine_kernel<<<dim3(H_DIM / 1024, T_NUM), 256, 0, stream>>>(
      ybuf, offsets, slot_e, slot_pos, slot_w, out);
}
